// Model_82411832476193
// MI455X (gfx1250) — matrix operand order repaired and checked by reading, not compiled, not run on hardware
//
#include <hip/hip_runtime.h>

#define N_NODES  100000
#define N_EDGES  1600000
#define N_GRAPHS 1000
#define HID      128
#define LAYERS   3
#define PROJ_IN  384

// LDS strides (in f16 elements), padded +8 so 16B-aligned chunk loads rotate banks
#define WS1 136   // for 128-deep K tiles
#define WSP 392   // for 384-deep K tiles (proj W1)

typedef __attribute__((ext_vector_type(16))) _Float16 v16h;
typedef __attribute__((ext_vector_type(8)))  _Float16 v8h;
typedef __attribute__((ext_vector_type(8)))  float    v8f;
typedef __attribute__((ext_vector_type(4)))  float    v4f;

__device__ __forceinline__ v8f zero8() {
  v8f z = {0.f,0.f,0.f,0.f,0.f,0.f,0.f,0.f};
  return z;
}

// ---------------------------------------------------------------------------
// A fragment (16x32 f16, MxK). CDNA5 layout:
//   lanes 0-15 : row = lane,    K = [k0..k0+7]   and [k0+16..k0+23]
//   lanes 16-31: row = lane-16, K = [k0+8..k0+15] and [k0+24..k0+31]
__device__ __forceinline__ v16h frag_a_lds(const _Float16* base, int stride, int k0) {
  const int lane = threadIdx.x & 31;
  const int r = lane & 15, hi = lane >> 4;
  const _Float16* p = base + r * stride + k0 + hi * 8;
  v8h c0 = *(const v8h*)(p);
  v8h c1 = *(const v8h*)(p + 16);
  v16h a;
#pragma unroll
  for (int i = 0; i < 8; ++i) { a[i] = c0[i]; a[i + 8] = c1[i]; }
  return a;
}

// B fragment (32x16 f16, KxN) read from transposed weight tile Wt[n][k]:
//   lanes 0-15 : col = lane,    K = [k0..k0+7]   and [k0+16..k0+23]
//   lanes 16-31: col = lane-16, K = [k0+8..k0+15] and [k0+24..k0+31]
__device__ __forceinline__ v16h frag_b_lds(const _Float16* wt, int stride, int ntile, int k0) {
  const int lane = threadIdx.x & 31;
  const int n = (lane & 15) + (ntile << 4), hi = lane >> 4;
  const _Float16* p = wt + n * stride + k0 + hi * 8;
  v8h c0 = *(const v8h*)(p);
  v8h c1 = *(const v8h*)(p + 16);
  v16h b;
#pragma unroll
  for (int i = 0; i < 8; ++i) { b[i] = c0[i]; b[i + 8] = c1[i]; }
  return b;
}

// A fragment straight from global f32 with fused agg = h + neigh, f32->f16
__device__ __forceinline__ v16h frag_a_agg(const float* __restrict__ h,
                                           const float* __restrict__ nb,
                                           long rowBase, int k0) {
  const int lane = threadIdx.x & 31;
  const int r = lane & 15, hi = lane >> 4;
  const long row = rowBase + r;
  v16h a;
  if (row < (long)N_NODES) {
    const float* ph = h  + row * HID + k0 + hi * 8;
    const float* pn = nb + row * HID + k0 + hi * 8;
    v4f h0 = *(const v4f*)(ph),      h1 = *(const v4f*)(ph + 4);
    v4f h2 = *(const v4f*)(ph + 16), h3 = *(const v4f*)(ph + 20);
    v4f n0 = *(const v4f*)(pn),      n1 = *(const v4f*)(pn + 4);
    v4f n2 = *(const v4f*)(pn + 16), n3 = *(const v4f*)(pn + 20);
#pragma unroll
    for (int i = 0; i < 4; ++i) {
      a[i]      = (_Float16)(h0[i] + n0[i]);
      a[i + 4]  = (_Float16)(h1[i] + n1[i]);
      a[i + 8]  = (_Float16)(h2[i] + n2[i]);
      a[i + 12] = (_Float16)(h3[i] + n3[i]);
    }
  } else {
#pragma unroll
    for (int i = 0; i < 16; ++i) a[i] = (_Float16)0.f;
  }
  return a;
}

// A fragment from a single global f32 matrix with leading dim ld
__device__ __forceinline__ v16h frag_a_g32(const float* __restrict__ g,
                                           long rowBase, int ld, int k0, long nrows) {
  const int lane = threadIdx.x & 31;
  const int r = lane & 15, hi = lane >> 4;
  const long row = rowBase + r;
  v16h a;
  if (row < nrows) {
    const float* p = g + row * ld + k0 + hi * 8;
    v4f x0 = *(const v4f*)(p),      x1 = *(const v4f*)(p + 4);
    v4f x2 = *(const v4f*)(p + 16), x3 = *(const v4f*)(p + 20);
#pragma unroll
    for (int i = 0; i < 4; ++i) {
      a[i]      = (_Float16)x0[i];
      a[i + 4]  = (_Float16)x1[i];
      a[i + 8]  = (_Float16)x2[i];
      a[i + 12] = (_Float16)x3[i];
    }
  } else {
#pragma unroll
    for (int i = 0; i < 16; ++i) a[i] = (_Float16)0.f;
  }
  return a;
}

// ---------------------------------------------------------------------------
__global__ void __launch_bounds__(256) zero_kernel(float* __restrict__ p, long n) {
  long i = ((long)blockIdx.x * blockDim.x + threadIdx.x) * 4;
  if (i + 3 < n) {
    v4f z = {0.f, 0.f, 0.f, 0.f};
    *(v4f*)(p + i) = z;
  } else {
    for (int k = 0; k < 4; ++k)
      if (i + k < n) p[i + k] = 0.f;
  }
}

// One wave32 per edge: lane l moves floats [4l, 4l+4) of the 128-wide row.
__global__ void __launch_bounds__(256) scatter_kernel(const float* __restrict__ h,
                                                      const int* __restrict__ esrc,
                                                      const int* __restrict__ edst,
                                                      float* __restrict__ neigh) {
  long idx = (long)blockIdx.x * blockDim.x + threadIdx.x;
  long e = idx >> 5;
  if (e >= (long)N_EDGES) return;
  const int lane = threadIdx.x & 31;
  const int s = esrc[e], d = edst[e];
  v4f v = *(const v4f*)(h + (long)s * HID + lane * 4);
  float* pd = neigh + (long)d * HID + lane * 4;
  atomicAdd(pd + 0, v[0]);
  atomicAdd(pd + 1, v[1]);
  atomicAdd(pd + 2, v[2]);
  atomicAdd(pd + 3, v[3]);
}

// ---------------------------------------------------------------------------
// Fused GIN layer: h_new = relu(relu((h + neigh) W1 + b1) W2 + b2)
// - neigh buffer `nio` is overwritten with h_new (rows are wave-private)
// - fused per-graph sum-pool via f32 atomics into pooled[g][layer*128 + n]
// 8 waves/block, 16 rows/wave => 128 rows/block.
__global__ void __launch_bounds__(256) gin_layer_kernel(
    const float* __restrict__ h, float* __restrict__ nio,
    const float* __restrict__ W1, const float* __restrict__ b1,
    const float* __restrict__ W2, const float* __restrict__ b2,
    const int* __restrict__ graph_ids, float* __restrict__ pooled, int layer) {
  extern __shared__ __align__(16) _Float16 smem[];
  _Float16* w1t = smem;                 // 128 x WS1  (Wt[n][k])
  _Float16* w2t = w1t + 128 * WS1;      // 128 x WS1
  _Float16* stg = w2t + 128 * WS1;      // 8 waves x 16 x WS1 (wave-private)
  __shared__ float sb1[HID], sb2[HID];

  const int tid = threadIdx.x;
  // cooperative transpose + f32->f16 of both weight matrices
  for (int i = tid; i < HID * HID; i += 256) {
    const int k = i >> 7, n = i & 127;      // i = k*128 + n
    w1t[n * WS1 + k] = (_Float16)W1[i];
    w2t[n * WS1 + k] = (_Float16)W2[i];
  }
  if (tid < HID) { sb1[tid] = b1[tid]; sb2[tid] = b2[tid]; }
  __syncthreads();

  const int wave = tid >> 5;
  const int lane = tid & 31;
  const int nloc = lane & 15, hi = lane >> 4;
  const long rowBase = ((long)blockIdx.x * 8 + wave) * 16;
  _Float16* mystg = stg + wave * 16 * WS1;

  v8f acc[8];
#pragma unroll
  for (int j = 0; j < 8; ++j) acc[j] = zero8();

  // GEMM1: (h + neigh)[16x128] @ W1[128x128]
#pragma unroll
  for (int ks = 0; ks < 4; ++ks) {
    v16h a = frag_a_agg(h, nio, rowBase, ks * 32);
#pragma unroll
    for (int j = 0; j < 8; ++j) {
      v16h b = frag_b_lds(w1t, WS1, j, ks * 32);
      acc[j] = __builtin_amdgcn_wmma_f32_16x16x32_f16(false, a, false, b,
                                                      (short)0, acc[j], false, false);
    }
  }

  // bias + relu -> wave-private f16 stage (D layout: row = r + 8*hi, col = nloc)
#pragma unroll
  for (int j = 0; j < 8; ++j) {
    const int n = (j << 4) + nloc;
    const float bv = sb1[n];
#pragma unroll
    for (int r = 0; r < 8; ++r) {
      float v = acc[j][r] + bv;
      v = v > 0.f ? v : 0.f;
      mystg[(r + 8 * hi) * WS1 + n] = (_Float16)v;
    }
  }

#pragma unroll
  for (int j = 0; j < 8; ++j) acc[j] = zero8();

  // GEMM2: stage[16x128] @ W2[128x128]
#pragma unroll
  for (int ks = 0; ks < 4; ++ks) {
    v16h a = frag_a_lds(mystg, WS1, ks * 32);
#pragma unroll
    for (int j = 0; j < 8; ++j) {
      v16h b = frag_b_lds(w2t, WS1, j, ks * 32);
      acc[j] = __builtin_amdgcn_wmma_f32_16x16x32_f16(false, a, false, b,
                                                      (short)0, acc[j], false, false);
    }
  }

  // bias + relu -> h_new (overwrite neigh buffer) + fused graph pooling
#pragma unroll
  for (int r = 0; r < 8; ++r) {
    const long row = rowBase + r + 8 * hi;
    if (row < (long)N_NODES) {
      const int g = graph_ids[row];
      float* orow = nio + row * HID;
      float* prow = pooled + (long)g * (LAYERS * HID) + layer * HID;
#pragma unroll
      for (int j = 0; j < 8; ++j) {
        const int n = (j << 4) + nloc;
        float v = acc[j][r] + sb2[n];
        v = v > 0.f ? v : 0.f;
        orow[n] = v;
        atomicAdd(prow + n, v);
      }
    }
  }
}

// ---------------------------------------------------------------------------
// Projection head: out = relu(pooled[1000x384] W1 + b1) W2 + b2
// 4 waves/block, 16 rows/wave => 64 rows/block.
__global__ void __launch_bounds__(128) proj_kernel(
    const float* __restrict__ pooled,
    const float* __restrict__ W1, const float* __restrict__ b1,
    const float* __restrict__ W2, const float* __restrict__ b2,
    float* __restrict__ out) {
  extern __shared__ __align__(16) _Float16 smem[];
  _Float16* w1t = smem;                 // 128 x WSP  (Wt[n][k], k in [0,384))
  _Float16* w2t = w1t + 128 * WSP;      // 128 x WS1
  _Float16* stg = w2t + 128 * WS1;      // 4 waves x 16 x WS1
  __shared__ float sb1[HID], sb2[HID];

  const int tid = threadIdx.x;
  for (int i = tid; i < PROJ_IN * HID; i += 128) {
    const int k = i >> 7, n = i & 127;
    w1t[n * WSP + k] = (_Float16)W1[i];
  }
  for (int i = tid; i < HID * HID; i += 128) {
    const int k = i >> 7, n = i & 127;
    w2t[n * WS1 + k] = (_Float16)W2[i];
  }
  if (tid < HID) { sb1[tid] = b1[tid]; sb2[tid] = b2[tid]; }
  __syncthreads();

  const int wave = tid >> 5;
  const int lane = tid & 31;
  const int nloc = lane & 15, hi = lane >> 4;
  const long rowBase = ((long)blockIdx.x * 4 + wave) * 16;
  _Float16* mystg = stg + wave * 16 * WS1;

  v8f acc[8];
#pragma unroll
  for (int j = 0; j < 8; ++j) acc[j] = zero8();

#pragma unroll
  for (int ks = 0; ks < PROJ_IN / 32; ++ks) {   // 12 K-steps
    v16h a = frag_a_g32(pooled, rowBase, PROJ_IN, ks * 32, (long)N_GRAPHS);
#pragma unroll
    for (int j = 0; j < 8; ++j) {
      v16h b = frag_b_lds(w1t, WSP, j, ks * 32);
      acc[j] = __builtin_amdgcn_wmma_f32_16x16x32_f16(false, a, false, b,
                                                      (short)0, acc[j], false, false);
    }
  }

#pragma unroll
  for (int j = 0; j < 8; ++j) {
    const int n = (j << 4) + nloc;
    const float bv = sb1[n];
#pragma unroll
    for (int r = 0; r < 8; ++r) {
      float v = acc[j][r] + bv;
      v = v > 0.f ? v : 0.f;
      mystg[(r + 8 * hi) * WS1 + n] = (_Float16)v;
    }
  }

#pragma unroll
  for (int j = 0; j < 8; ++j) acc[j] = zero8();

#pragma unroll
  for (int ks = 0; ks < 4; ++ks) {
    v16h a = frag_a_lds(mystg, WS1, ks * 32);
#pragma unroll
    for (int j = 0; j < 8; ++j) {
      v16h b = frag_b_lds(w2t, WS1, j, ks * 32);
      acc[j] = __builtin_amdgcn_wmma_f32_16x16x32_f16(false, a, false, b,
                                                      (short)0, acc[j], false, false);
    }
  }

#pragma unroll
  for (int r = 0; r < 8; ++r) {
    const long row = rowBase + r + 8 * hi;
    if (row < (long)N_GRAPHS) {
      float* orow = out + row * HID;
#pragma unroll
      for (int j = 0; j < 8; ++j) {
        const int n = (j << 4) + nloc;
        orow[n] = acc[j][r] + sb2[n];   // no relu on final linear
      }
    }
  }
}

// ---------------------------------------------------------------------------
extern "C" void kernel_launch(void* const* d_in, const int* in_sizes, int n_in,
                              void* d_out, int out_size, void* d_ws, size_t ws_size,
                              hipStream_t stream) {
  (void)in_sizes; (void)n_in; (void)out_size; (void)ws_size;
  const float* x         = (const float*)d_in[0];
  const int*   edge_src  = (const int*)d_in[1];
  const int*   edge_dst  = (const int*)d_in[2];
  const int*   graph_ids = (const int*)d_in[3];
  const float* gin_W1    = (const float*)d_in[4];
  const float* gin_b1    = (const float*)d_in[5];
  const float* gin_W2    = (const float*)d_in[6];
  const float* gin_b2    = (const float*)d_in[7];
  const float* proj_W1   = (const float*)d_in[8];
  const float* proj_b1   = (const float*)d_in[9];
  const float* proj_W2   = (const float*)d_in[10];
  const float* proj_b2   = (const float*)d_in[11];
  float* out = (float*)d_out;

  const long NH = (long)N_NODES * HID;
  float* buf0   = (float*)d_ws;                 // 100000*128 f32
  float* buf1   = buf0 + NH;                    // 100000*128 f32
  float* pooled = buf1 + NH;                    // 1000*384  f32

  const size_t ginLds  = (size_t)(2 * 128 * WS1 + 8 * 16 * WS1) * sizeof(_Float16);  // ~102 KB
  const size_t projLds = (size_t)(128 * WSP + 128 * WS1 + 4 * 16 * WS1) * sizeof(_Float16); // ~149 KB
  (void)hipFuncSetAttribute((const void*)gin_layer_kernel,
                            hipFuncAttributeMaxDynamicSharedMemorySize, (int)ginLds);
  (void)hipFuncSetAttribute((const void*)proj_kernel,
                            hipFuncAttributeMaxDynamicSharedMemorySize, (int)projLds);

  const int ginGrid     = (N_NODES + 127) / 128;                 // 782
  const long scatterThr = (long)N_EDGES * 32;
  const int scatterGrid = (int)((scatterThr + 255) / 256);       // 200000
  const int zeroGridNH  = (int)((NH / 4 + 255) / 256);
  const int zeroGridP   = (int)(((long)N_GRAPHS * PROJ_IN / 4 + 255) / 256);

  zero_kernel<<<zeroGridP, 256, 0, stream>>>(pooled, (long)N_GRAPHS * PROJ_IN);

  const float* hcur = x;
  float* nbufs[LAYERS] = {buf0, buf1, buf0};
  for (int l = 0; l < LAYERS; ++l) {
    float* nb = nbufs[l];
    zero_kernel<<<zeroGridNH, 256, 0, stream>>>(nb, NH);
    scatter_kernel<<<scatterGrid, 256, 0, stream>>>(hcur, edge_src, edge_dst, nb);
    gin_layer_kernel<<<ginGrid, 256, ginLds, stream>>>(
        hcur, nb,
        gin_W1 + (size_t)l * HID * HID, gin_b1 + (size_t)l * HID,
        gin_W2 + (size_t)l * HID * HID, gin_b2 + (size_t)l * HID,
        graph_ids, pooled, l);
    hcur = nb;
  }

  proj_kernel<<<(N_GRAPHS + 63) / 64, 128, projLds, stream>>>(
      pooled, proj_W1, proj_b1, proj_W2, proj_b2, out);
}
